// CausalSelfAttention_7773890806379
// MI455X (gfx1250) — matrix operand order repaired and checked by reading, not compiled, not run on hardware
//
#include <hip/hip_runtime.h>
#include <hip/hip_bf16.h>

typedef _Float16 f16;
typedef __attribute__((ext_vector_type(16))) _Float16 v16h;
typedef __attribute__((ext_vector_type(8)))  _Float16 v8h;
typedef __attribute__((ext_vector_type(8)))  float    v8f;
typedef __attribute__((ext_vector_type(4)))  float    v4f;

#define N_EMBED 768
#define N_HEAD  12
#define HDIM    64
#define BB      4
#define TT      2048
#define MTOT    (BB * TT)       // 8192 rows
#define C3      (3 * N_EMBED)   // 2304

// GEMM macro-tile: 128x128, K-step 32, 8 waves (2 M-halves x 4 N-quarters)
#define GM 128
#define GN 128
#define GK 32
#define LDT 72   // LDS row pitch (f16): 32 data + 40 pad = 144 B rows (16B aligned,
                 // 36-bank stride -> 16 distinct row-start banks)

__device__ __forceinline__ v8f wmma_f16f32(v16h a, v16h b, v8f c) {
  // v_wmma_f32_16x16x32_f16 : D = A(16x32,f16) x B(32x16,f16) + C(16x16,f32)
  return __builtin_amdgcn_wmma_f32_16x16x32_f16(false, a, false, b, (short)0, c,
                                                false, false);
}

// CDNA5 async tensor path: global -> LDS DMA, tracked by ASYNCcnt (ISA 15.18.3
// opcode 98). vdst = LDS byte address (low 32 bits of generic pointer, ISA 10.2).
__device__ __forceinline__ void async_copy_b128(const f16* g, unsigned lds_off) {
  asm volatile("global_load_async_to_lds_b128 %0, %1, off"
               :
               : "v"(lds_off), "v"(g)
               : "memory");
}
__device__ __forceinline__ void wait_async0() {
  asm volatile("s_wait_asynccnt 0x0" ::: "memory");
}

// A fragment (16x32 MxK, f16), source row-major, ld in elements.
// ISA 7.12.2: lane L -> row L%16; K = {klo..klo+7, klo+16..klo+23}, klo = 8*(L/16)
__device__ __forceinline__ v16h load_A_f16(const f16* __restrict__ tile, int ld) {
  const int lane = threadIdx.x & 31;
  const int r    = lane & 15;
  const int klo  = (lane >> 4) << 3;
  const f16* p = tile + (size_t)r * ld + klo;
  union { v16h v; v8h h[2]; } u;
  u.h[0] = *(const v8h*)(p);
  u.h[1] = *(const v8h*)(p + 16);
  return u.v;
}

// B fragment (32x16 KxN, f16) read from B^T (NxK row-major), ld in elements.
// ISA 7.12.4 pattern: lane L -> col L%16; K = {klo..klo+7, klo+16..klo+23}, klo = 8*(L/16)
__device__ __forceinline__ v16h load_Bt_f16(const f16* __restrict__ tileT, int ld) {
  const int lane = threadIdx.x & 31;
  const int n    = lane & 15;
  const int k0   = (lane >> 4) << 3;
  const f16* p = tileT + (size_t)n * ld + k0;
  union { v16h v; v8h h[2]; } u;
  u.h[0] = *(const v8h*)(p);
  u.h[1] = *(const v8h*)(p + 16);
  return u.v;
}

// ---------------- conversion kernels ----------------
__global__ void f32_to_f16_kernel(const float* __restrict__ src,
                                  f16* __restrict__ dst, int n8) {
  int i = blockIdx.x * blockDim.x + threadIdx.x;
  if (i < n8) {
    const v4f a = ((const v4f*)src)[2 * i];
    const v4f b = ((const v4f*)src)[2 * i + 1];
    v8h o;
#pragma unroll
    for (int e = 0; e < 4; ++e) { o[e] = (f16)a[e]; o[e + 4] = (f16)b[e]; }
    ((v8h*)dst)[i] = o;
  }
}

// src [K][N] (row-major) -> dst [N][K] f16 (i.e., B^T for WMMA B operand)
__global__ void transpose_f32_to_f16_kernel(const float* __restrict__ src,
                                            f16* __restrict__ dst, int K, int N) {
  int i = blockIdx.x * blockDim.x + threadIdx.x;
  if (i < K * N) {
    int kk = i / N;
    int nn = i - kk * N;
    dst[(size_t)nn * K + kk] = (f16)src[i];
  }
}

// ------------- shared 128x128 LDS-tiled, double-buffered GEMM core ----------
// Ag: tile origin of A (row-major, ld elems). Bg: tile origin of B^T (NxK).
// acc[i][j]: wave's 64x32 sub-tile = 4 M-frags x 2 N-frags.
// Tile copies use global_load_async_to_lds_b128 (ASYNCcnt double buffering).
__device__ __forceinline__ void gemm128_body(const f16* __restrict__ Ag,
                                             const f16* __restrict__ Bg,
                                             const int ld, const int kdim,
                                             v8f (&acc)[4][2]) {
  __shared__ f16 As[2][GM * LDT];
  __shared__ f16 Bs[2][GN * LDT];
  const int tid   = threadIdx.x;
  const int wid   = tid >> 5;
  const int wm    = wid & 1;    // M half (64 rows)
  const int wn    = wid >> 1;   // N quarter (32 cols)
  const int crow  = tid >> 1;        // copy row 0..127 (32 B = 2 x b128 per thread)
  const int ckoff = (tid & 1) << 4;  // 0 or 16 elements

  const unsigned ldsA0 = (unsigned)(size_t)(&As[0][crow * LDT + ckoff]);
  const unsigned ldsB0 = (unsigned)(size_t)(&Bs[0][crow * LDT + ckoff]);
  const unsigned bufStride = (unsigned)(GM * LDT * sizeof(f16));
  const f16* gA = Ag + (size_t)crow * ld + ckoff;
  const f16* gB = Bg + (size_t)crow * ld + ckoff;

  // prologue: async-fill buffer 0 with K-slab 0
  async_copy_b128(gA,     ldsA0);
  async_copy_b128(gA + 8, ldsA0 + 16);
  async_copy_b128(gB,     ldsB0);
  async_copy_b128(gB + 8, ldsB0 + 16);
  wait_async0();
  __syncthreads();

  int buf = 0;
  for (int k0 = 0; k0 < kdim; k0 += GK) {
    const bool more = (k0 + GK) < kdim;
    if (more) {   // async DMA of next K-slab into the other buffer
      const unsigned o = (unsigned)(buf ^ 1) * bufStride;
      async_copy_b128(gA + (k0 + GK),     ldsA0 + o);
      async_copy_b128(gA + (k0 + GK) + 8, ldsA0 + o + 16);
      async_copy_b128(gB + (k0 + GK),     ldsB0 + o);
      async_copy_b128(gB + (k0 + GK) + 8, ldsB0 + o + 16);
    }
    if (k0 + 2 * GK < kdim) {  // global_prefetch_b8 two slabs ahead
      __builtin_prefetch(gA + (k0 + 2 * GK), 0, 3);
      __builtin_prefetch(gB + (k0 + 2 * GK), 0, 3);
    }
    const f16* Ab = &As[buf][(wm * 64) * LDT];
    const f16* Bb = &Bs[buf][(wn * 32) * LDT];
    v16h af[4], bf[2];
#pragma unroll
    for (int i = 0; i < 4; ++i) af[i] = load_A_f16(Ab + (size_t)(16 * i) * LDT, LDT);
#pragma unroll
    for (int j = 0; j < 2; ++j) bf[j] = load_Bt_f16(Bb + (size_t)(16 * j) * LDT, LDT);
#pragma unroll
    for (int i = 0; i < 4; ++i)
#pragma unroll
      for (int j = 0; j < 2; ++j)
        acc[i][j] = wmma_f16f32(af[i], bf[j], acc[i][j]);
    if (more) wait_async0();   // own DMA done; barrier makes all waves' visible
    __syncthreads();
    buf ^= 1;
  }
}

// ---------------- QKV projection GEMM ----------------
__global__ __launch_bounds__(256)
void gemm_qkv_kernel(const f16* __restrict__ xh, const f16* __restrict__ wkt,
                     const float* __restrict__ bias,
                     f16* __restrict__ qo, f16* __restrict__ ko,
                     f16* __restrict__ vto) {
  const int mb = blockIdx.x << 7;
  const int nb = blockIdx.y << 7;
  v8f acc[4][2];
#pragma unroll
  for (int i = 0; i < 4; ++i)
#pragma unroll
    for (int j = 0; j < 2; ++j) acc[i][j] = {};
  gemm128_body(xh + (size_t)mb * N_EMBED, wkt + (size_t)nb * N_EMBED,
               N_EMBED, N_EMBED, acc);

  const int lane = threadIdx.x & 31;
  const int wid = threadIdx.x >> 5;
  const int wm = wid & 1, wn = wid >> 1;
#pragma unroll
  for (int i = 0; i < 4; ++i) {
#pragma unroll
    for (int j = 0; j < 2; ++j) {
      const int n0 = nb + wn * 32 + 16 * j;           // 16-col frag within one head
      const int jsec = n0 / N_EMBED;                  // 0=k, 1=q, 2=v (ref order)
      const int h = (n0 % N_EMBED) >> 6;
      const int d = (n0 & 63) + (lane & 15);
      const float bv = bias[n0 + (lane & 15)];
#pragma unroll
      for (int g = 0; g < 8; ++g) {
        const int m = mb + wm * 64 + 16 * i + g + ((lane >> 4) << 3);
        const int b = m >> 11, t = m & (TT - 1);
        const size_t bh = (size_t)(b * N_HEAD + h);
        const f16 val = (f16)(acc[i][j][g] + bv);
        if (jsec == 0)      ko[(bh * TT + t) * HDIM + d] = val;
        else if (jsec == 1) qo[(bh * TT + t) * HDIM + d] = val;
        else                vto[(bh * HDIM + d) * TT + t] = val; // v^T [D][T]
      }
    }
  }
}

// ---------------- output projection GEMM (fp32 out + bias) ------------------
__global__ __launch_bounds__(256)
void gemm_proj_kernel(const f16* __restrict__ aoh, const f16* __restrict__ wpt,
                      const float* __restrict__ bias, float* __restrict__ out) {
  const int mb = blockIdx.x << 7;
  const int nb = blockIdx.y << 7;
  v8f acc[4][2];
#pragma unroll
  for (int i = 0; i < 4; ++i)
#pragma unroll
    for (int j = 0; j < 2; ++j) acc[i][j] = {};
  gemm128_body(aoh + (size_t)mb * N_EMBED, wpt + (size_t)nb * N_EMBED,
               N_EMBED, N_EMBED, acc);

  const int lane = threadIdx.x & 31;
  const int wid = threadIdx.x >> 5;
  const int wm = wid & 1, wn = wid >> 1;
#pragma unroll
  for (int i = 0; i < 4; ++i) {
#pragma unroll
    for (int j = 0; j < 2; ++j) {
      const int n = nb + wn * 32 + 16 * j + (lane & 15);
      const float bv = bias[n];
#pragma unroll
      for (int g = 0; g < 8; ++g) {
        const int m = mb + wm * 64 + 16 * i + g + ((lane >> 4) << 3);
        out[(size_t)m * N_EMBED + n] = acc[i][j][g] + bv;
      }
    }
  }
}

// -------- flash attention: one wave = 32 q rows (two 16-row M-tiles) --------
__global__ __launch_bounds__(32)
void attn_fa_kernel(const f16* __restrict__ qh, const f16* __restrict__ kh,
                    const f16* __restrict__ vt, f16* __restrict__ aoh) {
  const int lane = threadIdx.x & 31;
  const int qt = blockIdx.x & 63;        // T/32 = 64 q tiles
  const int bh = blockIdx.x >> 6;        // b*12 + h
  const int h = bh % N_HEAD, b = bh / N_HEAD;
  const f16* qb = qh + (size_t)bh * TT * HDIM;
  const f16* kb = kh + (size_t)bh * TT * HDIM;
  const f16* vb = vt + (size_t)bh * HDIM * TT;
  const int qbase = qt << 5;

  v16h qA[2][2];
#pragma unroll
  for (int mt = 0; mt < 2; ++mt) {
    qA[mt][0] = load_A_f16(qb + (size_t)(qbase + 16 * mt) * HDIM + 0, HDIM);
    qA[mt][1] = load_A_f16(qb + (size_t)(qbase + 16 * mt) * HDIM + 32, HDIM);
  }

  v8f acc[2][4];
  float mrow[2][8], lrow[2][8], rfac[2][8];
#pragma unroll
  for (int mt = 0; mt < 2; ++mt) {
#pragma unroll
    for (int c = 0; c < 4; ++c) acc[mt][c] = {};
#pragma unroll
    for (int g = 0; g < 8; ++g) { mrow[mt][g] = -1e30f; lrow[mt][g] = 0.f; }
  }

  __shared__ f16 pl[2][16 * 32];

  const int kend = qbase + 32;
  for (int kbase = 0; kbase < kend; kbase += 32) {
    if (kbase + 32 < kend) {   // prefetch next k/v slab (cacheline pulls)
      __builtin_prefetch(kb + (size_t)(kbase + 32 + lane) * HDIM, 0, 3);
      __builtin_prefetch(vb + (size_t)(2 * lane) * TT + kbase + 32, 0, 3);
    }
    // shared k fragments: [t = 16-row k-subtile][dstep]
    v16h kB[2][2];
#pragma unroll
    for (int t = 0; t < 2; ++t) {
      kB[t][0] = load_Bt_f16(kb + (size_t)(kbase + 16 * t) * HDIM + 0, HDIM);
      kB[t][1] = load_Bt_f16(kb + (size_t)(kbase + 16 * t) * HDIM + 32, HDIM);
    }
    const int kc0 = kbase + (lane & 15);
#pragma unroll
    for (int mt = 0; mt < 2; ++mt) {
      v8f s0 = {}, s1 = {};
      s0 = wmma_f16f32(qA[mt][0], kB[0][0], s0);
      s0 = wmma_f16f32(qA[mt][1], kB[0][1], s0);
      s1 = wmma_f16f32(qA[mt][0], kB[1][0], s1);
      s1 = wmma_f16f32(qA[mt][1], kB[1][1], s1);
      const int rb0 = qbase + 16 * mt + ((lane >> 4) << 3);
#pragma unroll
      for (int g = 0; g < 8; ++g) {
        const int qi = rb0 + g;
        float a = s0[g] * 0.125f;            // 1/sqrt(64)
        float c = s1[g] * 0.125f;
        if (kc0 > qi)      a = -1e30f;       // causal mask
        if (kc0 + 16 > qi) c = -1e30f;
        float mx = fmaxf(a, c);
#pragma unroll
        for (int off = 8; off >= 1; off >>= 1)
          mx = fmaxf(mx, __shfl_xor(mx, off, 32));
        const float mn = fmaxf(mrow[mt][g], mx);
        const float r  = __expf(mrow[mt][g] - mn);
        const float e0 = __expf(a - mn);
        const float e1 = __expf(c - mn);
        float sum = e0 + e1;
#pragma unroll
        for (int off = 8; off >= 1; off >>= 1) sum += __shfl_xor(sum, off, 32);
        lrow[mt][g] = lrow[mt][g] * r + sum;
        mrow[mt][g] = mn;
        rfac[mt][g] = r;
        // stash p into LDS (C layout -> [16][32] row-major f16)
        const int col = lane & 15;
        const int rh  = ((lane >> 4) << 3) + g;
        pl[mt][rh * 32 + col]      = (f16)e0;
        pl[mt][rh * 32 + col + 16] = (f16)e1;
      }
    }
    __syncthreads();
    v16h pA[2];
#pragma unroll
    for (int mt = 0; mt < 2; ++mt) {
      const int r = lane & 15;
      const int klo = (lane >> 4) << 3;
#pragma unroll
      for (int i = 0; i < 8; ++i) {
        pA[mt][i]     = pl[mt][r * 32 + klo + i];
        pA[mt][i + 8] = pl[mt][r * 32 + klo + 16 + i];
      }
    }
    __syncthreads();
    // shared v fragments across both M-tiles
#pragma unroll
    for (int c = 0; c < 4; ++c) {
      v16h vB = load_Bt_f16(vb + (size_t)(16 * c) * TT + kbase, TT);
#pragma unroll
      for (int mt = 0; mt < 2; ++mt) {
        v8f a = acc[mt][c];
#pragma unroll
        for (int g = 0; g < 8; ++g) a[g] *= rfac[mt][g];
        acc[mt][c] = wmma_f16f32(pA[mt], vB, a);
      }
    }
  }
  // epilogue: normalize, store rows of [8192][768] f16
  const int col = lane & 15;
#pragma unroll
  for (int mt = 0; mt < 2; ++mt) {
#pragma unroll
    for (int g = 0; g < 8; ++g) {
      const int row = qbase + 16 * mt + ((lane >> 4) << 3) + g;
      const float inv = 1.0f / lrow[mt][g];
      const size_t base = ((size_t)(b * TT + row)) * N_EMBED + h * HDIM;
#pragma unroll
      for (int c = 0; c < 4; ++c)
        aoh[base + 16 * c + col] = (f16)(acc[mt][c][g] * inv);
    }
  }
}

// ---------------- host-side launcher ----------------
extern "C" void kernel_launch(void* const* d_in, const int* in_sizes, int n_in,
                              void* d_out, int out_size, void* d_ws, size_t ws_size,
                              hipStream_t stream) {
  (void)in_sizes; (void)n_in; (void)out_size; (void)ws_size;
  const float* x  = (const float*)d_in[0];
  const float* Wk = (const float*)d_in[1];
  const float* bk = (const float*)d_in[2];
  const float* Wp = (const float*)d_in[3];
  const float* bp = (const float*)d_in[4];
  float* out = (float*)d_out;

  char* ws = (char*)d_ws;
  size_t off = 0;
  f16* xh  = (f16*)(ws + off); off += (size_t)MTOT * N_EMBED * sizeof(f16);
  f16* wkt = (f16*)(ws + off); off += (size_t)C3 * N_EMBED * sizeof(f16);
  f16* wpt = (f16*)(ws + off); off += (size_t)N_EMBED * N_EMBED * sizeof(f16);
  f16* qh  = (f16*)(ws + off); off += (size_t)MTOT * N_EMBED * sizeof(f16);
  f16* kh  = (f16*)(ws + off); off += (size_t)MTOT * N_EMBED * sizeof(f16);
  f16* vtr = (f16*)(ws + off); off += (size_t)MTOT * N_EMBED * sizeof(f16);
  f16* aoh = (f16*)(ws + off); off += (size_t)MTOT * N_EMBED * sizeof(f16);

  const int nx8 = MTOT * N_EMBED / 8;
  f32_to_f16_kernel<<<(nx8 + 255) / 256, 256, 0, stream>>>(x, xh, nx8);
  transpose_f32_to_f16_kernel<<<(N_EMBED * C3 + 255) / 256, 256, 0, stream>>>(
      Wk, wkt, N_EMBED, C3);
  transpose_f32_to_f16_kernel<<<(N_EMBED * N_EMBED + 255) / 256, 256, 0, stream>>>(
      Wp, wpt, N_EMBED, N_EMBED);

  gemm_qkv_kernel<<<dim3(MTOT / GM, C3 / GN), 256, 0, stream>>>(xh, wkt, bk,
                                                                qh, kh, vtr);
  attn_fa_kernel<<<BB * N_HEAD * (TT / 32), 32, 0, stream>>>(qh, kh, vtr, aoh);
  gemm_proj_kernel<<<dim3(MTOT / GM, N_EMBED / GN), 256, 0, stream>>>(aoh, wpt,
                                                                      bp, out);
}
